// Block_34823594836120
// MI455X (gfx1250) — matrix operand order repaired and checked by reading, not compiled, not run on hardware
//
#include <hip/hip_runtime.h>

// ---------------------------------------------------------------------------
// Transformer block forward for MI455X (gfx1250, wave32, WMMA).
// fp32 in/out; bf16 WMMA (v_wmma_f32_16x16x32_bf16) for all matmuls.
// GEMM staging via Tensor Data Mover (tensor_load_to_lds, TENSORcnt) when the
// builtin exists, else per-lane GLOBAL_LOAD_ASYNC_TO_LDS_B128 (ASYNCcnt).
// ---------------------------------------------------------------------------

#define BB 4
#define TT 2048
#define CC 1024
#define HH 16
#define HD 64
#define MM (BB * TT)        // 8192 token rows
#define FF 4096             // 4*C

typedef __bf16 bf16_t;
typedef __attribute__((ext_vector_type(16))) __bf16 v16bf;
typedef __attribute__((ext_vector_type(8)))  __bf16 v8bf;
typedef __attribute__((ext_vector_type(8)))  float  v8f;
typedef __attribute__((ext_vector_type(4))) unsigned int u32x4;
typedef __attribute__((ext_vector_type(8))) int i32x8;
typedef __attribute__((ext_vector_type(4))) int i32x4;

#if defined(__has_builtin)
#if __has_builtin(__builtin_amdgcn_tensor_load_to_lds) && \
    __has_builtin(__builtin_amdgcn_s_wait_tensorcnt)
#define HAVE_TDM 1
#endif
#endif
#ifndef HAVE_TDM
#define HAVE_TDM 0
#endif

static __device__ __forceinline__ bf16_t f2bf(float f) {
  unsigned u = __builtin_bit_cast(unsigned, f);
  u += 0x7FFFu + ((u >> 16) & 1u);              // round-to-nearest-even
  unsigned short s = (unsigned short)(u >> 16);
  return __builtin_bit_cast(bf16_t, s);
}

static __device__ __forceinline__ v8f wmma_bf16(v16bf a, v16bf b, v8f c) {
  return __builtin_amdgcn_wmma_f32_16x16x32_bf16(
      false, a, false, b, (short)0, c, false, false);
}

static __device__ __forceinline__ void async_b128(unsigned lds, unsigned goff,
                                                  unsigned long long gbase) {
  asm volatile("global_load_async_to_lds_b128 %0, %1, %2"
               :: "v"(lds), "v"(goff), "s"(gbase) : "memory");
}
static __device__ __forceinline__ void wait_async0() {
  asm volatile("s_wait_asynccnt 0" ::: "memory");
}
static __device__ __forceinline__ unsigned lds_off(const void* p) {
  return (unsigned)(unsigned long long)p;   // LDS aperture: addr[31:0] = offset
}

#if HAVE_TDM
// TDM 2D tile load: tile_dim0=32 bf16 (64B rows), tile_dim1=rows, into LDS
// with hardware padding of 4 DWORDs every 16 DWORDs -> 40-element LDS stride.
// D# layout per CDNA5 ISA 8.3/8.4; 6-arg builtin form (extra i32x8 group).
static __device__ __forceinline__ void tdm_load_2d(unsigned lds_addr,
                                                   unsigned long long ga,
                                                   unsigned tile_rows,
                                                   unsigned stride_elems,
                                                   unsigned tensor_d0) {
  u32x4 g0;
  g0[0] = 1u;                                            // count=1, user D#
  g0[1] = lds_addr;                                      // lds_addr [63:32]
  g0[2] = (unsigned)(ga & 0xFFFFFFFFu);                  // global_addr lo
  g0[3] = (unsigned)((ga >> 32) & 0x01FFFFFFu) | (2u << 30);   // addr hi | type=2
  unsigned d1 = 1u << 20;                                // huge tensor_dim1
  i32x8 g1;
  g1[0] = (int)((1u << 16) |                 // data_size = 2 bytes
                (1u << 20) |                 // pad_enable
                (3u << 22) |                 // pad_interval: 16 DWORDs
                (3u << 25));                 // pad_amount:   4 DWORDs
  g1[1] = (int)((tensor_d0 & 0xFFFFu) << 16);            // tensor_dim0 lo16
  g1[2] = (int)(((tensor_d0 >> 16) & 0xFFFFu) |          // tensor_dim0 hi16
                ((d1 & 0xFFFFu) << 16));                 // tensor_dim1 lo16
  g1[3] = (int)(((d1 >> 16) & 0xFFFFu) | (32u << 16));   // dim1 hi | tile_dim0=32
  g1[4] = (int)(tile_rows & 0xFFFFu);                    // tile_dim1 (tile_dim2=0)
  g1[5] = (int)stride_elems;                             // tensor_dim0_stride lo32
  g1[6] = 0;                                             // stride hi | dim1_stride lo
  g1[7] = 0;
  i32x4 z4 = {0, 0, 0, 0};
  i32x8 z8 = {0, 0, 0, 0, 0, 0, 0, 0};
  __builtin_amdgcn_tensor_load_to_lds(g0, g1, z4, z4, z8, 0);
}
#endif

// A fragment: 16x32 bf16, row-major, stride lda elements (ISA 7.12.2).
static __device__ __forceinline__ v16bf load_a_frag(const bf16_t* A, int lda,
                                                    int row0, int k0, int lane) {
  int r  = row0 + (lane & 15);
  int kb = k0 + ((lane >> 4) << 3);
  const bf16_t* p = A + (size_t)r * lda + kb;
  v8bf lo = *(const v8bf*)(p);
  v8bf hi = *(const v8bf*)(p + 16);
  v16bf a;
#pragma unroll
  for (int i = 0; i < 8; ++i) { a[i] = lo[i]; a[i + 8] = hi[i]; }
  return a;
}

// B fragment from Bt stored [N][K] (K contiguous), via two 16B loads.
static __device__ __forceinline__ v16bf load_b_frag2(const bf16_t* Bt, int ldb,
                                                     int n0, int k0, int lane) {
  int n  = n0 + (lane & 15);
  int kb = k0 + ((lane >> 4) << 3);
  const bf16_t* p = Bt + (size_t)n * ldb + kb;
  v8bf lo = *(const v8bf*)(p);
  v8bf hi = *(const v8bf*)(p + 16);
  v16bf b;
#pragma unroll
  for (int i = 0; i < 8; ++i) { b[i] = lo[i]; b[i + 8] = hi[i]; }
  return b;
}

// ---------------------------------------------------------------------------
// Weight conversion: fp32 [K][N] row-major -> bf16 [N][K] transposed.
// ---------------------------------------------------------------------------
__global__ __launch_bounds__(256) void cvt_w_t(const float* __restrict__ W,
                                               bf16_t* __restrict__ Wt,
                                               int K, int N) {
  size_t i = (size_t)blockIdx.x * 256 + threadIdx.x;
  if (i >= (size_t)K * N) return;
  int n = (int)(i % N), k = (int)(i / N);
  Wt[(size_t)n * K + k] = f2bf(W[i]);
}

__global__ __launch_bounds__(256) void cvt_w_t_head(const float* __restrict__ W,
                                                    bf16_t* __restrict__ Wt,
                                                    int K, int N) {
  size_t i = (size_t)blockIdx.x * 256 + threadIdx.x;
  if (i >= (size_t)K * N) return;
  int n = (int)(i % N), k = (int)(i / N);
  int h = n >> 6, d = n & 63;
  Wt[(size_t)n * K + k] = f2bf(W[((size_t)h * K + k) * 64 + d]);
}

// ---------------------------------------------------------------------------
// LayerNorm over C=1024, one block per row, emits bf16.
// ---------------------------------------------------------------------------
__global__ __launch_bounds__(256) void layernorm_bf16(const float* __restrict__ x,
                                                      const float* __restrict__ g,
                                                      const float* __restrict__ b,
                                                      bf16_t* __restrict__ out,
                                                      int C) {
  int row = blockIdx.x;
  const float* xr = x + (size_t)row * C;
  float s = 0.f, s2 = 0.f;
  for (int c = threadIdx.x; c < C; c += 256) { float v = xr[c]; s += v; s2 += v * v; }
#pragma unroll
  for (int off = 16; off > 0; off >>= 1) {
    s  += __shfl_xor(s,  off, 32);
    s2 += __shfl_xor(s2, off, 32);
  }
  __shared__ float rs[8], rs2[8];
  int wv = threadIdx.x >> 5;
  if ((threadIdx.x & 31) == 0) { rs[wv] = s; rs2[wv] = s2; }
  __syncthreads();
  s = 0.f; s2 = 0.f;
#pragma unroll
  for (int i = 0; i < 8; ++i) { s += rs[i]; s2 += rs2[i]; }
  float mu  = s / C;
  float var = s2 / C - mu * mu;
  float inv = rsqrtf(var + 1e-5f);
  bf16_t* orow = out + (size_t)row * C;
  for (int c = threadIdx.x; c < C; c += 256)
    orow[c] = f2bf((xr[c] - mu) * inv * g[c] + b[c]);
}

// ---------------------------------------------------------------------------
// Block-tiled bf16 WMMA GEMM: out[m][n] = sum_k A[m][k]*Bt[n][k] (+bias)(+res)
// Block = 256 threads = 8 waves (4M x 2N); block tile 128(M) x 128(N);
// wave tile 32(M) x 64(N) = 8 WMMAs per 32-K chunk. K chunks double-buffered
// in LDS; staged by TDM (wave 0 only) or per-lane async copies.
// LDS row stride 40 elements (20 banks, matches TDM pad fields).
// ---------------------------------------------------------------------------
#define ASTR 40
__global__ __launch_bounds__(256) void gemm_bf16(const bf16_t* __restrict__ A,
                                                 const bf16_t* __restrict__ Bt,
                                                 const float* __restrict__ bias,
                                                 const float* __restrict__ resid,
                                                 float* __restrict__ outf,
                                                 bf16_t* __restrict__ outb,
                                                 int M, int N, int K, int relu) {
  __shared__ bf16_t Ab[2][128 * ASTR];
  __shared__ bf16_t Bb[2][128 * ASTR];

  int tid  = threadIdx.x;
  int lane = tid & 31, wv = tid >> 5;
  int tiles_n = N >> 7;
  int mt = blockIdx.x / tiles_n, nt = blockIdx.x - mt * tiles_n;
  int row0 = mt << 7, col0 = nt << 7;
  int mw = wv >> 1, nw = wv & 1;          // wave tile: 32 rows x 64 cols

#if !HAVE_TDM
  int r0  = tid >> 2, seg = (tid & 3) << 3;   // 128 rows x 4 segs = 2 per thread
  unsigned goff0 = (unsigned)((r0 * K + seg) * 2);
  unsigned goff1 = (unsigned)(((r0 + 64) * K + seg) * 2);
  unsigned a_l0[2], a_l1[2], b_l0[2], b_l1[2];
#pragma unroll
  for (int c = 0; c < 2; ++c) {
    a_l0[c] = lds_off(&Ab[c][r0 * ASTR + seg]);
    a_l1[c] = lds_off(&Ab[c][(r0 + 64) * ASTR + seg]);
    b_l0[c] = lds_off(&Bb[c][r0 * ASTR + seg]);
    b_l1[c] = lds_off(&Bb[c][(r0 + 64) * ASTR + seg]);
  }
#endif

  auto stage = [&](int kc, int buf) {
    unsigned long long ag = (unsigned long long)(A  + (size_t)row0 * K + kc * 32);
    unsigned long long bg = (unsigned long long)(Bt + (size_t)col0 * K + kc * 32);
#if HAVE_TDM
    if (wv == 0) {                       // one DMA per tile, wave 0 issues
      tdm_load_2d(lds_off(&Ab[buf][0]), ag, 128, (unsigned)K, (unsigned)K);
      tdm_load_2d(lds_off(&Bb[buf][0]), bg, 128, (unsigned)K, (unsigned)K);
    }
#else
    async_b128(a_l0[buf], goff0, ag);
    async_b128(a_l1[buf], goff1, ag);
    async_b128(b_l0[buf], goff0, bg);
    async_b128(b_l1[buf], goff1, bg);
#endif
  };
  auto wait_stage = [&]() {
#if HAVE_TDM
    if (wv == 0) __builtin_amdgcn_s_wait_tensorcnt(0);
#else
    wait_async0();
#endif
    __syncthreads();
  };

  int nk = K >> 5;
  stage(0, 0);
  wait_stage();

  v8f acc[2][4];
#pragma unroll
  for (int hh = 0; hh < 2; ++hh)
#pragma unroll
    for (int j = 0; j < 4; ++j) acc[hh][j] = v8f{};

  for (int kc = 0; kc < nk; ++kc) {
    int cur = kc & 1;
    if (kc + 1 < nk) stage(kc + 1, cur ^ 1);   // overlap copy with compute

    // load all fragments first so one dscnt wait covers the 8 WMMAs
    v16bf a0 = load_a_frag(&Ab[cur][0], ASTR, mw * 32,      0, lane);
    v16bf a1 = load_a_frag(&Ab[cur][0], ASTR, mw * 32 + 16, 0, lane);
    v16bf b0 = load_b_frag2(&Bb[cur][0], ASTR, nw * 64,      0, lane);
    v16bf b1 = load_b_frag2(&Bb[cur][0], ASTR, nw * 64 + 16, 0, lane);
    v16bf b2 = load_b_frag2(&Bb[cur][0], ASTR, nw * 64 + 32, 0, lane);
    v16bf b3 = load_b_frag2(&Bb[cur][0], ASTR, nw * 64 + 48, 0, lane);
    acc[0][0] = wmma_bf16(a0, b0, acc[0][0]);
    acc[0][1] = wmma_bf16(a0, b1, acc[0][1]);
    acc[0][2] = wmma_bf16(a0, b2, acc[0][2]);
    acc[0][3] = wmma_bf16(a0, b3, acc[0][3]);
    acc[1][0] = wmma_bf16(a1, b0, acc[1][0]);
    acc[1][1] = wmma_bf16(a1, b1, acc[1][1]);
    acc[1][2] = wmma_bf16(a1, b2, acc[1][2]);
    acc[1][3] = wmma_bf16(a1, b3, acc[1][3]);

    wait_stage();       // next chunk landed; all waves done reading cur
  }

  int col = lane & 15;
  int rb  = (lane >> 4) << 3;
#pragma unroll
  for (int hh = 0; hh < 2; ++hh)
#pragma unroll
    for (int j = 0; j < 4; ++j) {
      int n = col0 + nw * 64 + j * 16 + col;
      float bia = bias ? bias[n] : 0.f;
#pragma unroll
      for (int r = 0; r < 8; ++r) {
        int m = row0 + mw * 32 + hh * 16 + rb + r;
        float v = acc[hh][j][r] + bia;
        if (relu) v = v > 0.f ? v : 0.f;
        if (resid) v += resid[(size_t)m * N + n];
        if (outf) outf[(size_t)m * N + n] = v;
        if (outb) outb[(size_t)m * N + n] = f2bf(v);
      }
    }
}

// ---------------------------------------------------------------------------
// Flash attention. q,k,v,o stored bf16 [b*T + t][H*64] (head col offset h*64).
// Block = 256 threads = 8 waves, one (b,h), 128 query rows (16 per wave).
// K tile async-copied into LDS; V manually transposed into padded LDS.
// ---------------------------------------------------------------------------
__global__ __launch_bounds__(256) void flash_attn(const bf16_t* __restrict__ q,
                                                  const bf16_t* __restrict__ k,
                                                  const bf16_t* __restrict__ v,
                                                  bf16_t* __restrict__ o,
                                                  int B, int H, int T) {
  __shared__ bf16_t Ks[32 * 64];       // [key_t][d]
  __shared__ bf16_t Vt[64 * 48];       // [d][key_t], stride 48
  __shared__ bf16_t Ps[8][16 * 36];    // per-wave P scratch [row][k], stride 36

  int lane = threadIdx.x & 31;
  int wv   = threadIdx.x >> 5;
  int bh = blockIdx.x;
  int b = bh / H, h = bh - b * H;
  int C = H * 64;
  int qrow0 = blockIdx.y * 128 + wv * 16;

  const bf16_t* qb = q + (size_t)b * T * C + h * 64;
  const bf16_t* kb = k + (size_t)b * T * C + h * 64;
  const bf16_t* vb = v + (size_t)b * T * C + h * 64;

  v16bf qa0 = load_a_frag(qb, C, qrow0, 0,  lane);
  v16bf qa1 = load_a_frag(qb, C, qrow0, 32, lane);

  int tid = threadIdx.x;
  int rr = tid >> 3;                    // 0..31 key row in block
  int dd = (tid & 7) << 3;              // 0..56 d offset
  unsigned k_lds  = lds_off(&Ks[rr * 64 + dd]);
  unsigned k_goff = (unsigned)((rr * C + dd) * 2);

  v8f accO[4] = {v8f{}, v8f{}, v8f{}, v8f{}};
  float mrow[8], lrow[8];
#pragma unroll
  for (int r = 0; r < 8; ++r) { mrow[r] = -1e30f; lrow[r] = 0.f; }
  const float sc = 0.03125f * 1.44269504f;   // C^-0.5 * log2(e)

  for (int kt = 0; kt < T; kt += 32) {
    async_b128(k_lds, k_goff, (unsigned long long)(kb + (size_t)kt * C));
    {
      v8bf vvv = *(const v8bf*)(vb + (size_t)(kt + rr) * C + dd);
#pragma unroll
      for (int i = 0; i < 8; ++i) Vt[(dd + i) * 48 + rr] = vvv[i];
    }
    wait_async0();
    __syncthreads();

    v8f s0 = {}, s1 = {};
    s0 = wmma_bf16(qa0, load_b_frag2(Ks, 64, 0,  0,  lane), s0);
    s0 = wmma_bf16(qa1, load_b_frag2(Ks, 64, 0,  32, lane), s0);
    s1 = wmma_bf16(qa0, load_b_frag2(Ks, 64, 16, 0,  lane), s1);
    s1 = wmma_bf16(qa1, load_b_frag2(Ks, 64, 16, 32, lane), s1);

    float pm[8];
#pragma unroll
    for (int r = 0; r < 8; ++r) pm[r] = fmaxf(s0[r], s1[r]);
#pragma unroll
    for (int off = 1; off < 16; off <<= 1)
#pragma unroll
      for (int r = 0; r < 8; ++r) pm[r] = fmaxf(pm[r], __shfl_xor(pm[r], off, 32));

    float corr[8];
#pragma unroll
    for (int r = 0; r < 8; ++r) {
      float mn = fmaxf(mrow[r], pm[r]);
      corr[r] = exp2f((mrow[r] - mn) * sc);
      mrow[r] = mn;
    }
    float p0[8], p1[8], rsum[8];
#pragma unroll
    for (int r = 0; r < 8; ++r) {
      p0[r] = exp2f((s0[r] - mrow[r]) * sc);
      p1[r] = exp2f((s1[r] - mrow[r]) * sc);
      rsum[r] = p0[r] + p1[r];
    }
#pragma unroll
    for (int off = 1; off < 16; off <<= 1)
#pragma unroll
      for (int r = 0; r < 8; ++r) rsum[r] += __shfl_xor(rsum[r], off, 32);
#pragma unroll
    for (int r = 0; r < 8; ++r) lrow[r] = lrow[r] * corr[r] + rsum[r];
#pragma unroll
    for (int j = 0; j < 4; ++j)
#pragma unroll
      for (int r = 0; r < 8; ++r) accO[j][r] *= corr[r];

    bf16_t* ps = &Ps[wv][0];
    int col = lane & 15, rb2 = (lane >> 4) << 3;
#pragma unroll
    for (int r = 0; r < 8; ++r) {
      ps[(rb2 + r) * 36 + col]      = f2bf(p0[r]);
      ps[(rb2 + r) * 36 + col + 16] = f2bf(p1[r]);
    }
    asm volatile("s_wait_dscnt 0" ::: "memory");
    v16bf pa;
    {
      int prow = lane & 15, pkb = (lane >> 4) << 3;
      const bf16_t* pp = ps + prow * 36 + pkb;
#pragma unroll
      for (int i = 0; i < 8; ++i) { pa[i] = pp[i]; pa[i + 8] = pp[i + 16]; }
    }

#pragma unroll
    for (int j = 0; j < 4; ++j)
      accO[j] = wmma_bf16(pa, load_b_frag2(Vt, 48, j * 16, 0, lane), accO[j]);

    __syncthreads();
  }

  int col = lane & 15, rb2 = (lane >> 4) << 3;
  bf16_t* ob = o + (size_t)b * T * C + h * 64;
#pragma unroll
  for (int j = 0; j < 4; ++j)
#pragma unroll
    for (int r = 0; r < 8; ++r) {
      int t = qrow0 + rb2 + r;
      ob[(size_t)t * C + j * 16 + col] = f2bf(accO[j][r] / lrow[r]);
    }
}

// ---------------------------------------------------------------------------
// Host-side orchestration.
// ---------------------------------------------------------------------------
extern "C" void kernel_launch(void* const* d_in, const int* in_sizes, int n_in,
                              void* d_out, int out_size, void* d_ws, size_t ws_size,
                              hipStream_t stream) {
  (void)in_sizes; (void)n_in; (void)out_size; (void)ws_size;

  const float* x     = (const float*)d_in[0];
  const float* Wq    = (const float*)d_in[1];
  const float* Wk    = (const float*)d_in[2];
  const float* Wv    = (const float*)d_in[3];
  const float* Wproj = (const float*)d_in[4];
  const float* bproj = (const float*)d_in[5];
  const float* W1    = (const float*)d_in[6];
  const float* b1    = (const float*)d_in[7];
  const float* W2    = (const float*)d_in[8];
  const float* b2    = (const float*)d_in[9];
  const float* g1    = (const float*)d_in[10];
  const float* be1   = (const float*)d_in[11];
  const float* g2    = (const float*)d_in[12];
  const float* be2   = (const float*)d_in[13];

  char* p = (char*)d_ws;
  auto take = [&](size_t bytes) { char* q_ = p; p += (bytes + 255) & ~size_t(255); return q_; };

  bf16_t* h1   = (bf16_t*)take((size_t)MM * CC * 2);
  bf16_t* qbf  = (bf16_t*)take((size_t)MM * CC * 2);
  bf16_t* kbf  = (bf16_t*)take((size_t)MM * CC * 2);
  bf16_t* vbf  = (bf16_t*)take((size_t)MM * CC * 2);
  bf16_t* att  = (bf16_t*)take((size_t)MM * CC * 2);
  float*  x1   = (float*) take((size_t)MM * CC * 4);
  bf16_t* h2   = (bf16_t*)take((size_t)MM * CC * 2);
  bf16_t* a1   = (bf16_t*)take((size_t)MM * FF * 2);
  bf16_t* WqT  = (bf16_t*)take((size_t)CC * CC * 2);
  bf16_t* WkT  = (bf16_t*)take((size_t)CC * CC * 2);
  bf16_t* WvT  = (bf16_t*)take((size_t)CC * CC * 2);
  bf16_t* WpT  = (bf16_t*)take((size_t)CC * CC * 2);
  bf16_t* W1T  = (bf16_t*)take((size_t)CC * FF * 2);
  bf16_t* W2T  = (bf16_t*)take((size_t)FF * CC * 2);

  int cvb1 = (CC * CC + 255) / 256;
  int cvb4 = (CC * FF + 255) / 256;
  cvt_w_t_head<<<cvb1, 256, 0, stream>>>(Wq, WqT, CC, CC);
  cvt_w_t_head<<<cvb1, 256, 0, stream>>>(Wk, WkT, CC, CC);
  cvt_w_t_head<<<cvb1, 256, 0, stream>>>(Wv, WvT, CC, CC);
  cvt_w_t<<<cvb1, 256, 0, stream>>>(Wproj, WpT, CC, CC);
  cvt_w_t<<<cvb4, 256, 0, stream>>>(W1, W1T, CC, FF);
  cvt_w_t<<<cvb4, 256, 0, stream>>>(W2, W2T, FF, CC);

  layernorm_bf16<<<MM, 256, 0, stream>>>(x, g1, be1, h1, CC);

  // QKV GEMMs (M=8192, N=1024, K=1024): 64 x 8 block tiles of 128x128
  int gblk_c = (MM / 128) * (CC / 128);
  gemm_bf16<<<gblk_c, 256, 0, stream>>>(h1, WqT, nullptr, nullptr, nullptr, qbf, MM, CC, CC, 0);
  gemm_bf16<<<gblk_c, 256, 0, stream>>>(h1, WkT, nullptr, nullptr, nullptr, kbf, MM, CC, CC, 0);
  gemm_bf16<<<gblk_c, 256, 0, stream>>>(h1, WvT, nullptr, nullptr, nullptr, vbf, MM, CC, CC, 0);

  dim3 agrid(BB * HH, TT / 128);
  flash_attn<<<agrid, 256, 0, stream>>>(qbf, kbf, vbf, att, BB, HH, TT);

  // x1 = x + att @ Wproj + bproj
  gemm_bf16<<<gblk_c, 256, 0, stream>>>(att, WpT, bproj, x, x1, nullptr, MM, CC, CC, 0);

  layernorm_bf16<<<MM, 256, 0, stream>>>(x1, g2, be2, h2, CC);

  // FF1: a1 = relu(h2 @ W1 + b1)
  int gblk_f = (MM / 128) * (FF / 128);
  gemm_bf16<<<gblk_f, 256, 0, stream>>>(h2, W1T, b1, nullptr, nullptr, a1, MM, FF, CC, 1);

  // FF2: out = x1 + a1 @ W2 + b2
  gemm_bf16<<<gblk_c, 256, 0, stream>>>(a1, W2T, b2, x1, (float*)d_out, nullptr, MM, CC, FF, 0);
}
